// MoeLayer_84902913507408
// MI455X (gfx1250) — matrix operand order repaired and checked by reading, not compiled, not run on hardware
//
#include <hip/hip_runtime.h>
#include <hip/hip_bf16.h>
#include <math.h>

typedef __attribute__((ext_vector_type(16))) __bf16 v16bf;
typedef __attribute__((ext_vector_type(8)))  __bf16 v8bf;
typedef __attribute__((ext_vector_type(8)))  float  v8f;

constexpr int kB = 32;
constexpr int kT = 1024;
constexpr int kD = 512;
constexpr int kH = 2048;
constexpr int kE = 8;
constexpr int kTopK = 2;
constexpr int kTileT  = 32;    // token rows per workgroup (2 row-groups of 16)
constexpr int kHChunk = 256;   // hidden chunk staged in LDS

// ---------------------------------------------------------------------------
// Kernel 1: mean-pool over T, gate GEMV, top-2 + softmax. One block per batch.
// ---------------------------------------------------------------------------
__global__ void moe_gate_kernel(const float* __restrict__ x,
                                const float* __restrict__ gate_w,
                                const float* __restrict__ gate_b,
                                float* __restrict__ gate_out,   // [B,E] tail of d_out
                                int*   __restrict__ ridx,       // [B,2]
                                float* __restrict__ rwt) {      // [B,2]
    __shared__ float pool[kD];
    __shared__ float lg[kE];
    const int b   = blockIdx.x;
    const int tid = threadIdx.x;

    float s0 = 0.f, s1 = 0.f;
    for (int r = 0; r < kT; ++r) {
        const size_t base = ((size_t)b * kT + r) * kD;
        s0 += x[base + tid];
        s1 += x[base + tid + 256];
    }
    pool[tid]       = s0 * (1.0f / kT);
    pool[tid + 256] = s1 * (1.0f / kT);
    __syncthreads();

    if (tid < kE) {
        float acc = gate_b[tid];
        for (int d = 0; d < kD; ++d) acc += pool[d] * gate_w[d * kE + tid];
        lg[tid] = acc;
        gate_out[b * kE + tid] = acc;
    }
    __syncthreads();

    if (tid == 0) {
        int i0 = 0; float l0 = lg[0];
        for (int e = 1; e < kE; ++e) if (lg[e] > l0) { l0 = lg[e]; i0 = e; }
        int i1 = -1; float l1 = -1e30f;
        for (int e = 0; e < kE; ++e) if (e != i0 && lg[e] > l1) { l1 = lg[e]; i1 = e; }
        const float z  = expf(l1 - l0);       // l0 is the max
        const float p0 = 1.0f / (1.0f + z);
        const float p1 = z * p0;
        ridx[b * 2 + 0] = i0;  ridx[b * 2 + 1] = i1;
        rwt [b * 2 + 0] = p0;  rwt [b * 2 + 1] = p1;
    }
}

// ---------------------------------------------------------------------------
// Kernel 2: LDS-tiled transpose + fp32->bf16 (coalesced both sides).
//   w1 (E,D,H) -> w1t (E,H,D)   (GEMM1 B: row h, contiguous over k=d)
//   w2 (E,H,D) -> w2t (E,D,H)   (GEMM2 B: row d, contiguous over k=h)
// One 32x32 tile per block; [32][33] padding avoids LDS bank conflicts.
// ---------------------------------------------------------------------------
__global__ void moe_cvt_kernel(const float* __restrict__ w1,
                               const float* __restrict__ w2,
                               __bf16* __restrict__ w1t,
                               __bf16* __restrict__ w2t) {
    __shared__ float tile[32][33];
    const int nb1 = kE * (kD / 32) * (kH / 32);   // tiles in w1
    int bid = blockIdx.x;
    const float* src; __bf16* dst; int R, C;
    if (bid < nb1) { src = w1; dst = w1t; R = kD; C = kH; }
    else           { bid -= nb1; src = w2; dst = w2t; R = kH; C = kD; }
    const int tilesC    = C / 32;
    const int tilesPerE = (R / 32) * tilesC;
    const int e  = bid / tilesPerE;
    const int tr = bid % tilesPerE;
    const int r0 = (tr / tilesC) * 32;
    const int c0 = (tr % tilesC) * 32;
    const size_t base = (size_t)e * R * C;

    for (int i = threadIdx.x; i < 1024; i += 256) {
        const int rr = i >> 5, cc = i & 31;       // coalesced over C
        tile[rr][cc] = src[base + (size_t)(r0 + rr) * C + (c0 + cc)];
    }
    __syncthreads();
    for (int i = threadIdx.x; i < 1024; i += 256) {
        const int rr = i >> 5, cc = i & 31;       // coalesced over R in dst
        dst[base + (size_t)(c0 + rr) * R + (r0 + cc)] = (__bf16)tile[cc][rr];
    }
}

// ---------------------------------------------------------------------------
// A-fragment loader: 16-bit A 16x32 layout, contiguous k order (matches B):
//   lanes 0-15 : K = kk+0..15  (elems 0-15)
//   lanes 16-31: K = kk+16..31 (elems 0-15)
// ---------------------------------------------------------------------------
__device__ inline v16bf load_a_frag(const __bf16* row, int kk, int hi) {
    const v8bf lo  = *reinterpret_cast<const v8bf*>(row + kk + hi * 16);
    const v8bf hi8 = *reinterpret_cast<const v8bf*>(row + kk + hi * 16 + 8);
    return __builtin_shufflevector(lo, hi8, 0, 1, 2, 3, 4, 5, 6, 7,
                                   8, 9, 10, 11, 12, 13, 14, 15);
}

// ---------------------------------------------------------------------------
// Kernel 3: fused top-2 expert FFN for a 32-token tile of one batch.
// 8 wave32s = 2 row-groups x 4 col-groups. Routing weight pk is folded into
// the SiLU output so both experts accumulate into ONE set of output tiles:
//   out = sum_k (pk*silu(x@w1_k + b1_k)) @ w2_k  +  sum_k pk*b2_k  +  x
// ---------------------------------------------------------------------------
__global__ void __launch_bounds__(256)
moe_ffn_kernel(const float* __restrict__ x,
               const __bf16* __restrict__ w1t,   // (E,H,D) bf16
               const float*  __restrict__ b1,    // (E,H)
               const __bf16* __restrict__ w2t,   // (E,D,H) bf16
               const float*  __restrict__ b2,    // (E,D)
               const int*    __restrict__ ridx,  // (B,2)
               const float*  __restrict__ rwt,   // (B,2)
               float* __restrict__ out) {
    __shared__ __bf16 xs[kTileT * kD];       // 32 KB: x tile, row-major
    __shared__ __bf16 hs[kTileT * kHChunk];  // 16 KB: pk*silu(h) chunk

    const int b    = blockIdx.x >> 5;           // T/32 = 32 tiles per batch
    const int t0   = (blockIdx.x & 31) * kTileT;
    const int tid  = threadIdx.x;
    const int wid  = tid >> 5;
    const int lane = tid & 31;
    const int m    = lane & 15;                 // matrix row/col handled by lane
    const int hi   = lane >> 4;                 // half-wave selector
    const int rg   = wid >> 2;                  // row group 0..1 (16 rows each)
    const int cg   = wid & 3;                   // col group 0..3
    const int arow = rg * 16 + m;               // this lane's A row in xs/hs

    // Stage x tile as bf16 (coalesced fp32 reads).
    for (int i = tid; i < kTileT * kD; i += 256) {
        const int r = i >> 9, c = i & (kD - 1);
        xs[i] = (__bf16)x[((size_t)b * kT + t0 + r) * kD + c];
    }
    __syncthreads();

    const v8f vzero = {0.f, 0.f, 0.f, 0.f, 0.f, 0.f, 0.f, 0.f};
    v8f oacc[8];                                // 16 rows x 128 cols of output
    #pragma unroll
    for (int t = 0; t < 8; ++t) oacc[t] = vzero;

    for (int k = 0; k < kTopK; ++k) {
        const int   e  = ridx[b * kTopK + k];
        const float pk = rwt [b * kTopK + k];
        const __bf16* w1e = w1t + (size_t)e * kH * kD;
        const __bf16* w2e = w2t + (size_t)e * kD * kH;

        for (int hc = 0; hc < kH / kHChunk; ++hc) {
            // ---- GEMM1: h[32 x 256] = x_tile(32xD) @ w1[e] chunk ----
            // Wave covers rows rg*16..+16, cols cg*64..+64 (4 tiles).
            const __bf16* br1[4];
            #pragma unroll
            for (int t = 0; t < 4; ++t)
                br1[t] = w1e + (size_t)(hc * kHChunk + cg * 64 + t * 16 + m) * kD;

            v8f acc[4];
            #pragma unroll
            for (int t = 0; t < 4; ++t) acc[t] = vzero;

            #pragma unroll
            for (int kk = 0; kk < kD; kk += 32) {
                const v16bf a = load_a_frag(xs + arow * kD, kk, hi);  // 1 A load
                #pragma unroll
                for (int t = 0; t < 4; ++t) {                          // 4 WMMAs
                    const v16bf bf = *reinterpret_cast<const v16bf*>(br1[t] + kk + hi * 16);
                    acc[t] = __builtin_amdgcn_wmma_f32_16x16x32_bf16(
                        false, a, false, bf, (short)0, acc[t], false, false);
                }
            }

            __syncthreads();   // previous chunk's GEMM2 reads of hs are done
            #pragma unroll
            for (int t = 0; t < 4; ++t) {
                const int col  = cg * 64 + t * 16 + m;
                const float bias = b1[e * kH + hc * kHChunk + col];
                #pragma unroll
                for (int v = 0; v < 8; ++v) {
                    const float hv = acc[t][v] + bias;
                    const float sv = pk * hv / (1.0f + __expf(-hv));   // pk * SiLU
                    hs[(rg * 16 + v + 8 * hi) * kHChunk + col] = (__bf16)sv;
                }
            }
            __syncthreads();

            // ---- GEMM2: out[32 x D] += h_chunk(32x256) @ w2[e] slice ----
            // Wave covers rows rg*16..+16, cols cg*128..+128 (8 tiles).
            const __bf16* br2[8];
            #pragma unroll
            for (int t = 0; t < 8; ++t)
                br2[t] = w2e + (size_t)(cg * 128 + t * 16 + m) * kH + hc * kHChunk;

            #pragma unroll
            for (int kk = 0; kk < kHChunk; kk += 32) {
                const v16bf a = load_a_frag(hs + arow * kHChunk, kk, hi);
                #pragma unroll
                for (int t = 0; t < 8; ++t) {                          // 8 WMMAs
                    const v16bf bf = *reinterpret_cast<const v16bf*>(br2[t] + kk + hi * 16);
                    oacc[t] = __builtin_amdgcn_wmma_f32_16x16x32_bf16(
                        false, a, false, bf, (short)0, oacc[t], false, false);
                }
            }
        }
    }

    // Epilogue: + sum_k pk*b2_k + residual x. D-frag: row = v+8*hi, col = m.
    const int   e0 = ridx[b * 2 + 0], e1 = ridx[b * 2 + 1];
    const float p0 = rwt [b * 2 + 0], p1 = rwt [b * 2 + 1];
    #pragma unroll
    for (int t = 0; t < 8; ++t) {
        const int col = cg * 128 + t * 16 + m;
        const float bias = p0 * b2[e0 * kD + col] + p1 * b2[e1 * kD + col];
        #pragma unroll
        for (int v = 0; v < 8; ++v) {
            const int row = t0 + rg * 16 + v + 8 * hi;
            const size_t g = ((size_t)b * kT + row) * kD + col;
            out[g] = oacc[t][v] + bias + x[g];
        }
    }
}

// ---------------------------------------------------------------------------
extern "C" void kernel_launch(void* const* d_in, const int* in_sizes, int n_in,
                              void* d_out, int out_size, void* d_ws, size_t ws_size,
                              hipStream_t stream) {
    const float* x      = (const float*)d_in[0];
    const float* gate_w = (const float*)d_in[1];
    const float* gate_b = (const float*)d_in[2];
    const float* w1     = (const float*)d_in[3];
    const float* b1     = (const float*)d_in[4];
    const float* w2     = (const float*)d_in[5];
    const float* b2     = (const float*)d_in[6];

    float* outp     = (float*)d_out;
    float* gate_out = outp + (size_t)kB * kT * kD;   // tuple tail: gate_logits

    // Workspace layout: routing (1 KB) | w1t bf16 (16 MB) | w2t bf16 (16 MB)
    char*   wsb  = (char*)d_ws;
    int*    ridx = (int*)wsb;                                   // 64 ints
    float*  rwt  = (float*)(wsb + 256);                         // 64 floats
    __bf16* w1t  = (__bf16*)(wsb + 1024);
    __bf16* w2t  = (__bf16*)(wsb + 1024 + (size_t)kE * kD * kH * sizeof(__bf16));
    (void)in_sizes; (void)n_in; (void)out_size; (void)ws_size;

    moe_gate_kernel<<<kB, 256, 0, stream>>>(x, gate_w, gate_b, gate_out, ridx, rwt);

    const int cvt_blocks = 2 * kE * (kD / 32) * (kH / 32);      // 16384 tiles
    moe_cvt_kernel<<<cvt_blocks, 256, 0, stream>>>(w1, w2, w1t, w2t);

    moe_ffn_kernel<<<kB * (kT / kTileT), 256, 0, stream>>>(
        x, w1t, b1, w2t, b2, ridx, rwt, outp);
}
